// CustomMamba_28252294873791
// MI455X (gfx1250) — matrix operand order repaired and checked by reading, not compiled, not run on hardware
//
#include <hip/hip_runtime.h>
#include <hip/hip_bf16.h>

// ---------------- model constants ----------------
#define D_MODEL  768
#define D_INNER  1536
#define D_STATE  16
#define D_CONV   4
#define DT_RANK  48
#define N_LAYERS 24
#define BB 2
#define LL 1024
#define NTOK (BB * LL)          // 2048
#define EPSV 1e-5f
#define DBC_W 128               // padded dbc row stride (dt 0..47 | B 48..63 | C 64..79 | pad)

typedef __attribute__((ext_vector_type(16))) __bf16 v16bf;
typedef __attribute__((ext_vector_type(8)))  float  v8f;

__device__ __forceinline__ float sigmoidf_(float x) { return 1.f / (1.f + __expf(-x)); }
__device__ __forceinline__ float siluf_(float x)    { return x * sigmoidf_(x); }
__device__ __forceinline__ float softplusf_(float x){ return (x > 20.f) ? x : __logf(1.f + __expf(x)); }

// ---------------- f32 -> bf16 conversion ----------------
__global__ void f2bf_kernel(const float* __restrict__ src, __bf16* __restrict__ dst, int n) {
    int i = blockIdx.x * blockDim.x + threadIdx.x;
    if (i < n) dst[i] = (__bf16)src[i];
}

// pad W_x (1536x80 f32) -> (1536x128 bf16), zero cols 80..127
__global__ void pad_wx_kernel(const float* __restrict__ src, __bf16* __restrict__ dst) {
    int i = blockIdx.x * blockDim.x + threadIdx.x;
    if (i >= D_INNER * DBC_W) return;
    int row = i / DBC_W, col = i % DBC_W;
    dst[i] = (col < 80) ? (__bf16)src[row * 80 + col] : (__bf16)0.f;
}

// pad W_dt (48x1536 f32) -> (64x1536 bf16), zero rows 48..63
__global__ void pad_wdt_kernel(const float* __restrict__ src, __bf16* __restrict__ dst) {
    int i = blockIdx.x * blockDim.x + threadIdx.x;
    if (i >= 64 * D_INNER) return;
    int row = i / D_INNER, col = i % D_INNER;
    dst[i] = (row < DT_RANK) ? (__bf16)src[row * D_INNER + col] : (__bf16)0.f;
}

// pad dbc[:, :48] (stride DBC_W f32) -> (NTOK x 64 bf16), zero cols 48..63
__global__ void pad_dbc_kernel(const float* __restrict__ dbc, __bf16* __restrict__ dst) {
    int i = blockIdx.x * blockDim.x + threadIdx.x;
    if (i >= NTOK * 64) return;
    int row = i / 64, col = i % 64;
    dst[i] = (col < DT_RANK) ? (__bf16)dbc[(size_t)row * DBC_W + col] : (__bf16)0.f;
}

// concat(token_emb, times*w_time + b_time) -> bf16 [NTOK x 1536]
__global__ void embed_kernel(const float* __restrict__ emb, const float* __restrict__ times,
                             const float* __restrict__ wt, const float* __restrict__ bt,
                             __bf16* __restrict__ cat) {
    int idx = blockIdx.x * blockDim.x + threadIdx.x;
    if (idx >= NTOK * (2 * D_MODEL)) return;
    int tok = idx / (2 * D_MODEL), i = idx % (2 * D_MODEL);
    float v = (i < D_MODEL) ? emb[(size_t)tok * D_MODEL + i]
                            : times[tok] * wt[i - D_MODEL] + bt[i - D_MODEL];
    cat[idx] = (__bf16)v;
}

// ---------------- WMMA GEMM: C[M,N](f32) = A[M,K](bf16) @ B[K,N](bf16) ----------------
// Preconditions: M % 128 == 0, N % 64 == 0, K % 32 == 0, A/B 16B-aligned.
// 128x64 block tile, BK=32, double-buffered LDS (one barrier per K-step).
// 8 waves: wave (wm 0..3, wn 0..1) owns a 32x32 patch -> 4 WMMAs per K-step.
template<bool HAS_BIAS, bool HAS_RESID, bool SOFTPLUS>
__global__ __launch_bounds__(256) void gemm_wmma_kernel(
    const __bf16* __restrict__ A, const __bf16* __restrict__ Bm,
    float* __restrict__ C, const float* __restrict__ bias,
    const float* __restrict__ resid, int M, int N, int K)
{
    __shared__ __bf16 As[2][128][40];   // [buf][m][k], pitch 40 keeps 16B chunks aligned
    __shared__ __bf16 Bs[2][64][40];    // [buf][n][k] = B[k][n] transposed

    const int tid  = threadIdx.x;
    const int lane = tid & 31;
    const int wave = tid >> 5;
    const int half = lane >> 4;
    const int r    = lane & 15;
    const int wm   = wave & 3;
    const int wn   = wave >> 2;
    const int m0   = blockIdx.y * 128;
    const int n0   = blockIdx.x * 64;

    // per-thread staging coordinates (8-elem = 16B chunks)
    const int am  = tid >> 2;            // A rows am and am+64
    const int akc = (tid & 3) * 8;
    const int bkk = tid >> 3;
    const int bnc = (tid & 7) * 8;

    uint4 aR0, aR1, bR;
    auto loadG = [&](int k0) {
        aR0 = *reinterpret_cast<const uint4*>(A + (size_t)(m0 + am) * K + k0 + akc);
        aR1 = *reinterpret_cast<const uint4*>(A + (size_t)(m0 + 64 + am) * K + k0 + akc);
        bR  = *reinterpret_cast<const uint4*>(Bm + (size_t)(k0 + bkk) * N + n0 + bnc);
    };
    auto storeL = [&](int buf) {
        *reinterpret_cast<uint4*>(&As[buf][am][akc])      = aR0;
        *reinterpret_cast<uint4*>(&As[buf][64 + am][akc]) = aR1;
        const __bf16* rv = reinterpret_cast<const __bf16*>(&bR);
#pragma unroll
        for (int j = 0; j < 8; ++j) Bs[buf][bnc + j][bkk] = rv[j];
    };

    v8f acc00 = {}, acc01 = {}, acc10 = {}, acc11 = {};

    loadG(0);
    storeL(0);
    __syncthreads();

    const int nt = K >> 5;
    for (int kt = 0; kt < nt; ++kt) {
        const int cur = kt & 1;
        const bool more = (kt + 1) < nt;
        if (more) loadG((kt + 1) << 5);   // prefetch next tile while computing

        // --- fragments (contiguous k = 16*half + s for A and B) ---
        v16bf a0, a1, b0, b1;
        const __bf16* ap0 = &As[cur][wm * 32 + r][half * 16];
        const __bf16* ap1 = &As[cur][wm * 32 + 16 + r][half * 16];
        const __bf16* bp0 = &Bs[cur][wn * 32 + r][half * 16];
        const __bf16* bp1 = &Bs[cur][wn * 32 + 16 + r][half * 16];
#pragma unroll
        for (int j = 0; j < 8; ++j) {
            a0[j] = ap0[j]; a0[8 + j] = ap0[8 + j];
            a1[j] = ap1[j]; a1[8 + j] = ap1[8 + j];
        }
#pragma unroll
        for (int j = 0; j < 16; ++j) { b0[j] = bp0[j]; b1[j] = bp1[j]; }

        acc00 = __builtin_amdgcn_wmma_f32_16x16x32_bf16(false, a0, false, b0, (short)0, acc00, false, false);
        acc01 = __builtin_amdgcn_wmma_f32_16x16x32_bf16(false, a0, false, b1, (short)0, acc01, false, false);
        acc10 = __builtin_amdgcn_wmma_f32_16x16x32_bf16(false, a1, false, b0, (short)0, acc10, false, false);
        acc11 = __builtin_amdgcn_wmma_f32_16x16x32_bf16(false, a1, false, b1, (short)0, acc11, false, false);

        if (more) storeL(cur ^ 1);        // fill the other buffer
        __syncthreads();                   // one barrier per K-step
    }

    // --- epilogue (compile-time specialized): row = v + half*8, col = lane&15 ---
#pragma unroll
    for (int mi = 0; mi < 2; ++mi) {
        const v8f& aA = (mi == 0) ? acc00 : acc10;
        const v8f& aB = (mi == 0) ? acc01 : acc11;
        int rowBase = m0 + wm * 32 + mi * 16 + half * 8;
#pragma unroll
        for (int v = 0; v < 8; ++v) {
            int row = rowBase + v;
#pragma unroll
            for (int ni = 0; ni < 2; ++ni) {
                int col = n0 + wn * 32 + ni * 16 + r;
                float val = (ni == 0) ? aA[v] : aB[v];
                if constexpr (HAS_BIAS)  val += bias[col];
                if constexpr (HAS_RESID) val += resid[(size_t)row * N + col];
                if constexpr (SOFTPLUS)  val = softplusf_(val);
                C[(size_t)row * N + col] = val;
            }
        }
    }
}

// ---------------- RMSNorm over D_MODEL, one block per token ----------------
__global__ __launch_bounds__(256) void rmsnorm_kernel(
    const float* __restrict__ X, const float* __restrict__ w,
    __bf16* __restrict__ outB, float* __restrict__ outF)
{
    __shared__ float red[256];
    int tok = blockIdx.x;
    const float* xp = X + (size_t)tok * D_MODEL;
    float ss = 0.f;
    for (int i = threadIdx.x; i < D_MODEL; i += 256) { float v = xp[i]; ss += v * v; }
    red[threadIdx.x] = ss; __syncthreads();
    for (int s = 128; s > 0; s >>= 1) {
        if (threadIdx.x < s) red[threadIdx.x] += red[threadIdx.x + s];
        __syncthreads();
    }
    float scale = rsqrtf(red[0] / (float)D_MODEL + EPSV);
    for (int i = threadIdx.x; i < D_MODEL; i += 256) {
        float v = xp[i] * scale * w[i];
        if (outB) outB[(size_t)tok * D_MODEL + i] = (__bf16)v;
        if (outF) outF[(size_t)tok * D_MODEL + i] = v;
    }
}

// ---------------- depthwise causal conv(4) + SiLU + mask, write bf16 u ----------------
__global__ void conv_silu_kernel(const float* __restrict__ UZ, const float* __restrict__ cw,
                                 const float* __restrict__ cb, const float* __restrict__ mask,
                                 __bf16* __restrict__ UC)
{
    int idx = blockIdx.x * blockDim.x + threadIdx.x;
    if (idx >= NTOK * D_INNER) return;
    int d   = idx % D_INNER;
    int tok = idx / D_INNER;
    int b   = tok >> 10;
    int l   = tok & 1023;
    float acc = cb[d];
#pragma unroll
    for (int k = 0; k < D_CONV; ++k) {
        int li = l - (D_CONV - 1) + k;
        if (li >= 0) {
            int t2 = (b << 10) + li;
            acc += cw[d * D_CONV + k] * UZ[(size_t)t2 * (2 * D_INNER) + d] * mask[t2];
        }
    }
    UC[idx] = (__bf16)(siluf_(acc) * mask[tok]);
}

// ---------------- selective scan: thread per (b,d), 16-state recurrence ----------------
#define TCH 64
__global__ __launch_bounds__(256) void scan_kernel(
    const float* __restrict__ DT, const float* __restrict__ DBC,
    const __bf16* __restrict__ U, const float* __restrict__ Alog,
    float* __restrict__ Y)
{
    __shared__ float Bc[TCH][D_STATE];
    __shared__ float Cc[TCH][D_STATE];
    const int blkPerB = D_INNER / 256;                 // 6
    int b = blockIdx.x / blkPerB;
    int d = (blockIdx.x % blkPerB) * 256 + threadIdx.x;

    float Arow[D_STATE];
#pragma unroll
    for (int n = 0; n < D_STATE; ++n) Arow[n] = -__expf(Alog[(size_t)d * D_STATE + n]);
    float h[D_STATE];
#pragma unroll
    for (int n = 0; n < D_STATE; ++n) h[n] = 0.f;

    for (int l0 = 0; l0 < LL; l0 += TCH) {
        // stage B/C rows for this chunk: TCH*32 floats, 8 per thread
#pragma unroll
        for (int i = 0; i < 8; ++i) {
            int idx = threadIdx.x + i * 256;
            int t = idx >> 5, c = idx & 31;
            float v = DBC[(size_t)(b * LL + l0 + t) * DBC_W + DT_RANK + c];
            if (c < D_STATE) Bc[t][c] = v; else Cc[t][c - D_STATE] = v;
        }
        __syncthreads();
        for (int t = 0; t < TCH; ++t) {
            size_t tok = (size_t)(b * LL + l0 + t);
            float dtv = DT[tok * D_INNER + d];
            float uv  = (float)U[tok * D_INNER + d];
            float dbu = dtv * uv;
            float y = 0.f;
#pragma unroll
            for (int n = 0; n < D_STATE; ++n) {
                float dA = __expf(dtv * Arow[n]);
                h[n] = dA * h[n] + dbu * Bc[t][n];
                y += h[n] * Cc[t][n];
            }
            Y[tok * D_INNER + d] = y;
        }
        __syncthreads();
    }
}

// ---------------- y = (y + u*D_skip) * silu(z), write bf16 ----------------
__global__ void gate_kernel(const float* __restrict__ Y, const __bf16* __restrict__ U,
                            const float* __restrict__ UZ, const float* __restrict__ Dk,
                            __bf16* __restrict__ YB)
{
    int idx = blockIdx.x * blockDim.x + threadIdx.x;
    if (idx >= NTOK * D_INNER) return;
    int d = idx % D_INNER, tok = idx / D_INNER;
    float z = UZ[(size_t)tok * (2 * D_INNER) + D_INNER + d];
    float val = (Y[idx] + (float)U[idx] * Dk[d]) * siluf_(z);
    YB[idx] = (__bf16)val;
}

// ---------------- masked mean pooling over L ----------------
__global__ void pool_kernel(const float* __restrict__ XN, const float* __restrict__ mask,
                            float* __restrict__ pooled)
{
    int idx = blockIdx.x * blockDim.x + threadIdx.x;
    if (idx >= BB * D_MODEL) return;
    int b = idx / D_MODEL, i = idx % D_MODEL;
    float s = 0.f, ms = 0.f;
    for (int l = 0; l < LL; ++l) {
        float m = mask[b * LL + l];
        s  += XN[(size_t)(b * LL + l) * D_MODEL + i] * m;
        ms += m;
    }
    pooled[idx] = s / fmaxf(ms, 1e-9f);
}

// ---------------- classifier head ----------------
__global__ void head1_kernel(const float* __restrict__ pooled, const float* __restrict__ W1,
                             const float* __restrict__ b1, float* __restrict__ H1)
{
    int idx = blockIdx.x * blockDim.x + threadIdx.x;
    if (idx >= BB * (D_MODEL / 2)) return;
    int b = idx / (D_MODEL / 2), j = idx % (D_MODEL / 2);
    float s = b1[j];
    for (int i = 0; i < D_MODEL; ++i) s += pooled[b * D_MODEL + i] * W1[i * (D_MODEL / 2) + j];
    H1[idx] = fmaxf(s, 0.f);
}

__global__ void head2_kernel(const float* __restrict__ H1, const float* __restrict__ W2,
                             const float* __restrict__ b2, float* __restrict__ out)
{
    int b = threadIdx.x;
    if (b < BB) {
        float s = b2[0];
        for (int j = 0; j < D_MODEL / 2; ++j) s += H1[b * (D_MODEL / 2) + j] * W2[j];
        out[b] = s;
    }
}

// ---------------- host orchestration ----------------
extern "C" void kernel_launch(void* const* d_in, const int* in_sizes, int n_in,
                              void* d_out, int out_size, void* d_ws, size_t ws_size,
                              hipStream_t stream) {
    (void)in_sizes; (void)n_in; (void)out_size; (void)ws_size;

    const float* emb      = (const float*)d_in[0];
    const float* times    = (const float*)d_in[1];
    const float* mask     = (const float*)d_in[2];
    const float* w_time   = (const float*)d_in[3];
    const float* b_time   = (const float*)d_in[4];
    const float* W_comb   = (const float*)d_in[5];
    const float* b_comb   = (const float*)d_in[6];
    const float* norm_w   = (const float*)d_in[7];
    const float* W_in     = (const float*)d_in[8];
    const float* conv_w   = (const float*)d_in[9];
    const float* conv_b   = (const float*)d_in[10];
    const float* W_x      = (const float*)d_in[11];
    const float* W_dt     = (const float*)d_in[12];
    const float* b_dt     = (const float*)d_in[13];
    const float* A_log    = (const float*)d_in[14];
    const float* D_skip   = (const float*)d_in[15];
    const float* W_out    = (const float*)d_in[16];
    const float* norm_f_w = (const float*)d_in[17];
    const float* W_c1     = (const float*)d_in[18];
    const float* b_c1     = (const float*)d_in[19];
    const float* W_c2     = (const float*)d_in[20];
    const float* b_c2     = (const float*)d_in[21];

    char* ws = (char*)d_ws;
    size_t off = 0;
    auto alloc = [&](size_t bytes) -> char* {
        char* p = ws + off;
        off = (off + bytes + 255) & ~(size_t)255;
        return p;
    };

    float*  X      = (float*) alloc((size_t)NTOK * D_MODEL * 4);
    __bf16* XN_BF  = (__bf16*)alloc((size_t)NTOK * D_MODEL * 2);
    float*  UZ     = (float*) alloc((size_t)NTOK * 2 * D_INNER * 4);  // also reused as XNF
    __bf16* UC_BF  = (__bf16*)alloc((size_t)NTOK * D_INNER * 2);
    float*  DBC    = (float*) alloc((size_t)NTOK * DBC_W * 4);
    __bf16* DTB_BF = (__bf16*)alloc((size_t)NTOK * 64 * 2);
    float*  DT     = (float*) alloc((size_t)NTOK * D_INNER * 4);
    float*  Y      = (float*) alloc((size_t)NTOK * D_INNER * 4);
    __bf16* YBF    = (__bf16*)alloc((size_t)NTOK * D_INNER * 2);      // also CAT buffer
    __bf16* WBF    = (__bf16*)alloc((size_t)D_MODEL * 4 * D_INNER * 2); // max weight (W_in)
    float*  POOL   = (float*) alloc((size_t)BB * D_MODEL * 4);
    float*  H1     = (float*) alloc((size_t)BB * (D_MODEL / 2) * 4);
    float*  XNF    = UZ; // final normed f32 reuses UZ region

    dim3 blk(256);
    dim3 gEw((NTOK * D_INNER + 255) / 256);
    dim3 gCat((NTOK * 2 * D_MODEL + 255) / 256);

    // x = concat(emb, time_emb) @ W_comb + b_comb
    embed_kernel<<<gCat, blk, 0, stream>>>(emb, times, w_time, b_time, YBF /*cat*/);
    f2bf_kernel<<<(2 * D_MODEL * D_MODEL + 255) / 256, blk, 0, stream>>>(W_comb, WBF, 2 * D_MODEL * D_MODEL);
    gemm_wmma_kernel<true, false, false><<<dim3(D_MODEL / 64, NTOK / 128), blk, 0, stream>>>(
        YBF, WBF, X, b_comb, nullptr, NTOK, D_MODEL, 2 * D_MODEL);

    for (int l = 0; l < N_LAYERS; ++l) {
        // RMSNorm -> bf16
        rmsnorm_kernel<<<NTOK, blk, 0, stream>>>(X, norm_w + (size_t)l * D_MODEL, XN_BF, nullptr);

        // uz = h @ W_in   (2048 x 3072 x 768)
        f2bf_kernel<<<(D_MODEL * 2 * D_INNER + 255) / 256, blk, 0, stream>>>(
            W_in + (size_t)l * D_MODEL * 2 * D_INNER, WBF, D_MODEL * 2 * D_INNER);
        gemm_wmma_kernel<false, false, false><<<dim3(2 * D_INNER / 64, NTOK / 128), blk, 0, stream>>>(
            XN_BF, WBF, UZ, nullptr, nullptr, NTOK, 2 * D_INNER, D_MODEL);

        // depthwise conv + silu + mask
        conv_silu_kernel<<<gEw, blk, 0, stream>>>(
            UZ, conv_w + (size_t)l * D_INNER * D_CONV, conv_b + (size_t)l * D_INNER, mask, UC_BF);

        // dbc = u @ W_x  (2048 x 128pad x 1536)
        pad_wx_kernel<<<(D_INNER * DBC_W + 255) / 256, blk, 0, stream>>>(
            W_x + (size_t)l * D_INNER * 80, WBF);
        gemm_wmma_kernel<false, false, false><<<dim3(DBC_W / 64, NTOK / 128), blk, 0, stream>>>(
            UC_BF, WBF, DBC, nullptr, nullptr, NTOK, DBC_W, D_INNER);

        // dt = softplus(dbc[:, :48] @ W_dt + b_dt)  (K padded 48 -> 64)
        pad_dbc_kernel<<<(NTOK * 64 + 255) / 256, blk, 0, stream>>>(DBC, DTB_BF);
        pad_wdt_kernel<<<(64 * D_INNER + 255) / 256, blk, 0, stream>>>(
            W_dt + (size_t)l * DT_RANK * D_INNER, WBF);
        gemm_wmma_kernel<true, false, true><<<dim3(D_INNER / 64, NTOK / 128), blk, 0, stream>>>(
            DTB_BF, WBF, DT, b_dt + (size_t)l * D_INNER, nullptr, NTOK, D_INNER, 64);

        // selective scan
        scan_kernel<<<BB * (D_INNER / 256), blk, 0, stream>>>(
            DT, DBC, UC_BF, A_log + (size_t)l * D_INNER * D_STATE, Y);

        // gate: (y + u*D) * silu(z) -> bf16
        gate_kernel<<<gEw, blk, 0, stream>>>(Y, UC_BF, UZ, D_skip + (size_t)l * D_INNER, YBF);

        // x = x + y @ W_out  (2048 x 768 x 1536, residual)
        f2bf_kernel<<<(D_INNER * D_MODEL + 255) / 256, blk, 0, stream>>>(
            W_out + (size_t)l * D_INNER * D_MODEL, WBF, D_INNER * D_MODEL);
        gemm_wmma_kernel<false, true, false><<<dim3(D_MODEL / 64, NTOK / 128), blk, 0, stream>>>(
            YBF, WBF, X, nullptr, X, NTOK, D_MODEL, D_INNER);
    }

    // final norm (f32), pooling, head
    rmsnorm_kernel<<<NTOK, blk, 0, stream>>>(X, norm_f_w, nullptr, XNF);
    pool_kernel<<<(BB * D_MODEL + 255) / 256, blk, 0, stream>>>(XNF, mask, POOL);
    head1_kernel<<<(BB * (D_MODEL / 2) + 255) / 256, blk, 0, stream>>>(POOL, W_c1, b_c1, H1);
    head2_kernel<<<1, 64, 0, stream>>>(H1, W_c2, b_c2, (float*)d_out);
}
